// Transformer_45097156608239
// MI455X (gfx1250) — matrix operand order repaired and checked by reading, not compiled, not run on hardware
//
#include <hip/hip_runtime.h>
#include <hip/hip_bf16.h>
#include <math.h>

typedef __attribute__((ext_vector_type(16))) __bf16 v16bf;
typedef __attribute__((ext_vector_type(8)))  float  v8f;
typedef __attribute__((ext_vector_type(4)))  unsigned int u32x4;

#define BB 2
#define TT 1024
#define CC 768
#define HH 12
#define DD 64
#define VV 32000
#define LL 6
#define FF (4*CC)     /* 3072 */
#define MM (BB*TT)    /* 2048 */

// ---------------------------------------------------------------- helpers ---

__device__ __forceinline__ unsigned short f2bfu(float f) {
  // f32 -> bf16 round-to-nearest-even (bit pattern)
  union { float f; unsigned u; } a; a.f = f;
  unsigned r = a.u + 0x7FFFu + ((a.u >> 16) & 1u);
  return (unsigned short)(r >> 16);
}

// A-matrix 16x32 bf16 fragment (cdna5 layout): lane<16 row=lr holds
// K = {h*8..h*8+7} in v0-3 and {16+h*8..} in v4-7.  rowp = &row[k0]
__device__ __forceinline__ v16bf frag_a16(const unsigned short* rowp, int half) {
  union { v16bf v; u32x4 u[2]; } f;
  f.u[0] = *(const u32x4*)(rowp + half * 8);
  f.u[1] = *(const u32x4*)(rowp + 16 + half * 8);
  return f.v;
}
// B-matrix 32x16 bf16 fragment (same layout as A): col=lr holds
// K = {h*8..h*8+7} in v0-3 and {16+h*8..} in v4-7.  colp = &col[k0] (k contiguous)
__device__ __forceinline__ v16bf frag_b16(const unsigned short* colp, int half) {
  union { v16bf v; u32x4 u[2]; } f;
  f.u[0] = *(const u32x4*)(colp + half * 8);
  f.u[1] = *(const u32x4*)(colp + 16 + half * 8);
  return f.v;
}

// async copy of one 16B chunk: global (bf16) -> LDS, tracked by ASYNCcnt
__device__ __forceinline__ void async_g2l_b128(unsigned lds_addr, const void* gp) {
  asm volatile("global_load_async_to_lds_b128 %0, %1, off"
               :: "v"(lds_addr), "v"((unsigned long long)(size_t)gp)
               : "memory");
}
__device__ __forceinline__ void wait_async0() {
  asm volatile("s_wait_asynccnt 0" ::: "memory");
}

// ------------------------------------------------------------- embedding ---

__global__ void embed_k(const int* __restrict__ x, const float* __restrict__ tok,
                        const float* __restrict__ pos, float* __restrict__ h) {
  int e = blockIdx.x * blockDim.x + threadIdx.x;     // over MM*CC
  if (e >= MM * CC) return;
  int c = e % CC, row = e / CC, t = row % TT;
  h[e] = tok[(size_t)x[row] * CC + c] + pos[(size_t)t * CC + c];
}

// ----------------------------------------------- layernorm (f32 -> bf16) ---

__global__ __launch_bounds__(256) void ln_k(const float* __restrict__ in,
                                            const float* __restrict__ g,
                                            const float* __restrict__ b,
                                            unsigned short* __restrict__ out) {
  int row = blockIdx.x, tid = threadIdx.x;
  __shared__ float red[256];
  float loc[3];
  float s = 0.f;
#pragma unroll
  for (int i = 0; i < 3; ++i) { loc[i] = in[(size_t)row * CC + tid + i * 256]; s += loc[i]; }
  red[tid] = s; __syncthreads();
  for (int o = 128; o > 0; o >>= 1) { if (tid < o) red[tid] += red[tid + o]; __syncthreads(); }
  float mean = red[0] * (1.0f / CC);
  __syncthreads();
  float s2 = 0.f;
#pragma unroll
  for (int i = 0; i < 3; ++i) { float d = loc[i] - mean; s2 += d * d; }
  red[tid] = s2; __syncthreads();
  for (int o = 128; o > 0; o >>= 1) { if (tid < o) red[tid] += red[tid + o]; __syncthreads(); }
  float inv = rsqrtf(red[0] * (1.0f / CC) + 1e-5f);
#pragma unroll
  for (int i = 0; i < 3; ++i) {
    int c = tid + i * 256;
    out[(size_t)row * CC + c] = f2bfu((loc[i] - mean) * inv * g[c] + b[c]);
  }
}

// -------------------------------------------- generic bf16-WMMA GEMM -------
// A: bf16 row-major [M,K]  (staged to LDS with GLOBAL_LOAD_ASYNC_TO_LDS_B128)
// B: f32, element (k,n) at (n/bHead)*K*bHead + k*bHead + n%bHead
//    (bHead==N => row-major, bHead==64 => (H,C,D) QKV weight layout);
//    converted f32->bf16 on the fly (weights are single-use: no pre-convert).
// Out: f32 (OutF) or bf16 (OutB).  M%128==N%128==K%32==0.

#define GBM 128
#define GBN 128
#define GBK 32
#define LSTR 40   /* bf16 elems per LDS row: 32 data + 8 pad (80B, conflict-free) */

__global__ __launch_bounds__(256) void gemm_k(const unsigned short* __restrict__ A,
                                              const float* __restrict__ Bm,
                                              float* __restrict__ OutF,
                                              unsigned short* __restrict__ OutB,
                                              const float* __restrict__ bias,
                                              const float* __restrict__ resid,
                                              int M, int N, int K, int bHead, int doRelu) {
  __shared__ __align__(16) unsigned short As[GBM * LSTR];
  __shared__ __align__(16) unsigned short Bs[GBN * LSTR];
  int tid = threadIdx.x;
  int lane = tid & 31, wid = tid >> 5;
  int half = lane >> 4, lr = lane & 15;
  int warpM = wid & 1, warpN = wid >> 1;          // 2x4 waves -> 128x128 block
  int m0 = blockIdx.y * GBM, n0 = blockIdx.x * GBN;

  v8f acc[4][2];
#pragma unroll
  for (int i = 0; i < 4; ++i)
#pragma unroll
    for (int j = 0; j < 2; ++j)
#pragma unroll
      for (int r = 0; r < 8; ++r) acc[i][j][r] = 0.f;

  for (int k0 = 0; k0 < K; k0 += GBK) {
    // ---- A tile: async bf16 copy, 128 rows x 32 cols = 512 x 16B chunks ----
    {
      const unsigned short* ga = A + (size_t)m0 * K + k0;
#pragma unroll
      for (int p = 0; p < 2; ++p) {
        int e = tid + p * 256;                     // 512 chunks total
        int rowl = e >> 2, ch = e & 3;
        unsigned lds = (unsigned)(unsigned long long)(const void*)&As[rowl * LSTR + ch * 8];
        async_g2l_b128(lds, ga + (size_t)rowl * K + ch * 8);
      }
    }
    // ---- B tile: f32 load + cvt + ds store (transposed to [n][k]) ----------
    for (int e = tid; e < GBN * GBK; e += 256) {
      int nl = e & 127, kl = e >> 7;
      int n = n0 + nl, k = k0 + kl;
      Bs[nl * LSTR + kl] =
          f2bfu(Bm[(size_t)(n / bHead) * K * bHead + (size_t)k * bHead + (n % bHead)]);
    }
    // prefetch next weight tile into GL2 while we compute this one
    if (k0 + GBK < K) {
      int n = n0 + (tid & 127), k = k0 + GBK + (tid >> 7);
      __builtin_prefetch(&Bm[(size_t)(n / bHead) * K * bHead + (size_t)k * bHead + (n % bHead)], 0, 0);
    }
    wait_async0();
    __syncthreads();

    v16bf af[4], bfr[2];
#pragma unroll
    for (int i = 0; i < 4; ++i)
      af[i] = frag_a16(&As[(warpM * 64 + i * 16 + lr) * LSTR], half);
#pragma unroll
    for (int j = 0; j < 2; ++j)
      bfr[j] = frag_b16(&Bs[(warpN * 32 + j * 16 + lr) * LSTR], half);
#pragma unroll
    for (int i = 0; i < 4; ++i)
#pragma unroll
      for (int j = 0; j < 2; ++j)
        acc[i][j] = __builtin_amdgcn_wmma_f32_16x16x32_bf16(
            false, af[i], false, bfr[j], (short)0, acc[i][j], false, false);
    __syncthreads();
  }

#pragma unroll
  for (int i = 0; i < 4; ++i)
#pragma unroll
    for (int j = 0; j < 2; ++j)
#pragma unroll
      for (int r = 0; r < 8; ++r) {
        int row = m0 + warpM * 64 + i * 16 + half * 8 + r;
        int col = n0 + warpN * 32 + j * 16 + lr;
        float val = acc[i][j][r];
        if (bias)  val += bias[col];
        if (resid) val += resid[(size_t)row * N + col];
        if (doRelu) val = fmaxf(val, 0.f);
        if (OutB) OutB[(size_t)row * N + col] = f2bfu(val);
        else      OutF[(size_t)row * N + col] = val;
      }
}

// --------------------------- V transpose: [M,C] bf16 -> [B,H,D,T] bf16 -----

__global__ void vtrans_k(const unsigned short* __restrict__ v,
                         unsigned short* __restrict__ vtb) {
  int e = blockIdx.x * blockDim.x + threadIdx.x;   // over MM*CC
  if (e >= MM * CC) return;
  int c = e % CC, row = e / CC, b = row / TT, t = row % TT;
  int hh = c / DD, d = c % DD;
  vtb[(((size_t)b * HH + hh) * DD + d) * TT + t] = v[e];
}

// ------------------------------------------ flash attention (WMMA, wave32) -
// Q,K: bf16 row-major [M, C] (head slice = col offset h*D); Vt: bf16 [B,H,D,T]
// grid.x = B*H*(T/64); block = 128 (4 waves, 16 query rows per wave)

__global__ __launch_bounds__(128) void attn_k(const unsigned short* __restrict__ Qb,
                                              const unsigned short* __restrict__ Kb,
                                              const unsigned short* __restrict__ Vtb,
                                              unsigned short* __restrict__ O, float scale) {
  __shared__ __align__(16) unsigned short pl[4][16][16];   // per-wave P tile
  int lane = threadIdx.x & 31, wid = threadIdx.x >> 5;
  int half = lane >> 4, lr = lane & 15;
  const int qgroups = TT / 64;
  int bh = blockIdx.x / qgroups;
  int qbase = (blockIdx.x % qgroups) * 64 + wid * 16;
  int b = bh / HH, hh = bh % HH;
  const unsigned short* Q  = Qb + (size_t)b * TT * CC + hh * DD;
  const unsigned short* Kp = Kb + (size_t)b * TT * CC + hh * DD;
  const unsigned short* Vt = Vtb + (size_t)bh * DD * TT;

  v16bf aq[2];
  aq[0] = frag_a16(Q + (size_t)(qbase + lr) * CC, half);
  aq[1] = frag_a16(Q + (size_t)(qbase + lr) * CC + 32, half);

  float mrow[8], lsum[8];
  v8f oacc[4];
#pragma unroll
  for (int r = 0; r < 8; ++r) { mrow[r] = -3.0e38f; lsum[r] = 0.f; }
#pragma unroll
  for (int j = 0; j < 4; ++j)
#pragma unroll
    for (int r = 0; r < 8; ++r) oacc[j][r] = 0.f;

  int ktmax = (qbase + 15) >> 4;
  for (int kt = 0; kt <= ktmax; ++kt) {
    const unsigned short* krow = Kp + (size_t)(kt * 16 + lr) * CC;
    v16bf bk0 = frag_b16(krow, half);
    v16bf bk1 = frag_b16(krow + 32, half);
    v8f s;
#pragma unroll
    for (int r = 0; r < 8; ++r) s[r] = 0.f;
    s = __builtin_amdgcn_wmma_f32_16x16x32_bf16(false, aq[0], false, bk0, (short)0, s, false, false);
    s = __builtin_amdgcn_wmma_f32_16x16x32_bf16(false, aq[1], false, bk1, (short)0, s, false, false);

    float p[8], alpha[8];
    int scol = kt * 16 + lr;
#pragma unroll
    for (int r = 0; r < 8; ++r) {
      int trow = qbase + half * 8 + r;
      float sv = s[r] * scale;
      if (scol > trow) sv = -3.0e38f;               // causal mask
      float rm = sv;
      rm = fmaxf(rm, __shfl_xor(rm, 1, 16));
      rm = fmaxf(rm, __shfl_xor(rm, 2, 16));
      rm = fmaxf(rm, __shfl_xor(rm, 4, 16));
      rm = fmaxf(rm, __shfl_xor(rm, 8, 16));
      float mnew = fmaxf(mrow[r], rm);
      alpha[r] = __expf(mrow[r] - mnew);
      mrow[r] = mnew;
      float pv = __expf(sv - mnew);
      p[r] = pv;
      float ps = pv;
      ps += __shfl_xor(ps, 1, 16);
      ps += __shfl_xor(ps, 2, 16);
      ps += __shfl_xor(ps, 4, 16);
      ps += __shfl_xor(ps, 8, 16);
      lsum[r] = lsum[r] * alpha[r] + ps;
    }
    // C-layout (col per lane) -> A-layout (row per lane) via per-wave LDS tile
#pragma unroll
    for (int r = 0; r < 8; ++r) pl[wid][half * 8 + r][lr] = f2bfu(p[r]);
    asm volatile("s_wait_dscnt 0" ::: "memory");
    union { v16bf v; u32x4 u[2]; } fp;
    u32x4 z = {0u, 0u, 0u, 0u};
    fp.u[0] = half ? z : *(const u32x4*)(&pl[wid][lr][0]);
    fp.u[1] = half ? z : *(const u32x4*)(&pl[wid][lr][8]);   // K=16..31 of P (half 1) are zero
    int kb2 = kt * 16 + half * 16;
    if (kb2 + 16 > TT) kb2 -= 16;                   // clamp (zero-P half anyway)
#pragma unroll
    for (int j = 0; j < 4; ++j) {
#pragma unroll
      for (int r = 0; r < 8; ++r) oacc[j][r] *= alpha[r];
      union { v16bf v; u32x4 u[2]; } fv;
      const unsigned short* vp = Vt + (size_t)(j * 16 + lr) * TT + kb2;
      fv.u[0] = *(const u32x4*)(vp);
      fv.u[1] = *(const u32x4*)(vp + 8);
      oacc[j] = __builtin_amdgcn_wmma_f32_16x16x32_bf16(
          false, fp.v, false, fv.v, (short)0, oacc[j], false, false);
    }
  }

#pragma unroll
  for (int r = 0; r < 8; ++r) {
    float inv = 1.0f / lsum[r];
    int trow = qbase + half * 8 + r;
#pragma unroll
    for (int j = 0; j < 4; ++j)
      O[(size_t)(b * TT + trow) * CC + hh * DD + j * 16 + lr] = f2bfu(oacc[j][r] * inv);
  }
}

// ------------------------------------------------------------------ host ---

extern "C" void kernel_launch(void* const* d_in, const int* in_sizes, int n_in,
                              void* d_out, int out_size, void* d_ws, size_t ws_size,
                              hipStream_t stream) {
  (void)in_sizes; (void)n_in; (void)out_size; (void)ws_size;
  const int*   x    = (const int*)  d_in[0];
  const float* tok  = (const float*)d_in[1];
  const float* pos  = (const float*)d_in[2];
  const float* Wq   = (const float*)d_in[3];
  const float* Wk   = (const float*)d_in[4];
  const float* Wv   = (const float*)d_in[5];
  const float* Wo   = (const float*)d_in[6];
  const float* bo   = (const float*)d_in[7];
  const float* ln1g = (const float*)d_in[8];
  const float* ln1b = (const float*)d_in[9];
  const float* ln2g = (const float*)d_in[10];
  const float* ln2b = (const float*)d_in[11];
  const float* W1   = (const float*)d_in[12];
  const float* b1   = (const float*)d_in[13];
  const float* W2   = (const float*)d_in[14];
  const float* b2   = (const float*)d_in[15];
  const float* lnfg = (const float*)d_in[16];
  const float* lnfb = (const float*)d_in[17];
  const float* Wh   = (const float*)d_in[18];
  const float* bh   = (const float*)d_in[19];
  float* out = (float*)d_out;

  char* ws = (char*)d_ws;
  const size_t F32ACT = (size_t)MM * CC * 4;        // 6,291,456 B
  const size_t BFACT  = (size_t)MM * CC * 2;        // 3,145,728 B
  float*          h    = (float*)(ws);
  unsigned short* Ax   = (unsigned short*)(ws + F32ACT);
  unsigned short* qb16 = (unsigned short*)(ws + F32ACT + 1 * BFACT);
  unsigned short* kb16 = (unsigned short*)(ws + F32ACT + 2 * BFACT);
  unsigned short* v16  = (unsigned short*)(ws + F32ACT + 3 * BFACT);
  unsigned short* vtb  = (unsigned short*)(ws + F32ACT + 4 * BFACT);
  unsigned short* Ob16 = (unsigned short*)(ws + F32ACT + 5 * BFACT);
  unsigned short* HID  = (unsigned short*)(ws + F32ACT + 6 * BFACT);  // MM*FF*2

  float scale = 1.0f / sqrtf((float)CC);            // reference scales by C^-0.5

  dim3 b256(256), b128(128);
  embed_k<<<(MM * CC) / 256, b256, 0, stream>>>(x, tok, pos, h);
  for (int l = 0; l < LL; ++l) {
    ln_k<<<MM, b256, 0, stream>>>(h, ln1g + (size_t)l * CC, ln1b + (size_t)l * CC, Ax);
    gemm_k<<<dim3(CC / 128, MM / 128), b256, 0, stream>>>(
        Ax, Wq + (size_t)l * HH * CC * DD, nullptr, qb16, nullptr, nullptr, MM, CC, CC, DD, 0);
    gemm_k<<<dim3(CC / 128, MM / 128), b256, 0, stream>>>(
        Ax, Wk + (size_t)l * HH * CC * DD, nullptr, kb16, nullptr, nullptr, MM, CC, CC, DD, 0);
    gemm_k<<<dim3(CC / 128, MM / 128), b256, 0, stream>>>(
        Ax, Wv + (size_t)l * HH * CC * DD, nullptr, v16, nullptr, nullptr, MM, CC, CC, DD, 0);
    vtrans_k<<<(MM * CC) / 256, b256, 0, stream>>>(v16, vtb);
    attn_k<<<BB * HH * (TT / 64), b128, 0, stream>>>(qb16, kb16, vtb, Ob16, scale);
    gemm_k<<<dim3(CC / 128, MM / 128), b256, 0, stream>>>(
        Ob16, Wo + (size_t)l * CC * CC, h, nullptr, bo + (size_t)l * CC, h, MM, CC, CC, CC, 0);
    ln_k<<<MM, b256, 0, stream>>>(h, ln2g + (size_t)l * CC, ln2b + (size_t)l * CC, Ax);
    gemm_k<<<dim3(FF / 128, MM / 128), b256, 0, stream>>>(
        Ax, W1 + (size_t)l * CC * FF, nullptr, HID, b1 + (size_t)l * FF, nullptr, MM, FF, CC, FF, 1);
    gemm_k<<<dim3(CC / 128, MM / 128), b256, 0, stream>>>(
        HID, W2 + (size_t)l * FF * CC, h, nullptr, b2 + (size_t)l * CC, h, MM, CC, FF, CC, 0);
  }
  ln_k<<<MM, b256, 0, stream>>>(h, lnfg, lnfb, Ax);
  gemm_k<<<dim3(VV / 128, MM / 128), b256, 0, stream>>>(
      Ax, Wh, out, nullptr, bh, nullptr, MM, VV, CC, VV, 0);
}
